// stgnn_gat_improve_21199958573703
// MI455X (gfx1250) — compile-verified, matrix operand order checked by reading, not run on hardware
//
#include <hip/hip_runtime.h>

typedef __attribute__((ext_vector_type(16))) _Float16 v16h;
typedef __attribute__((ext_vector_type(8)))  float    v8f;

#define DEVINL __device__ __forceinline__

// K/channel index held by (lane-group g, half h) in a 16-bit A/B WMMA fragment
DEVINL int kmap(int g, int h) { return (h < 8) ? (8 * g + h) : (8 + 8 * g + h); }
// channel -> LDS slot permutation [0..7,16..23,8..15,24..31] (involution)
DEVINL int chperm(int c) {
  int blk = (c >> 3) & 3;
  int nb  = (blk == 1) ? 2 : ((blk == 2) ? 1 : blk);
  return (c & 7) | (nb << 3);
}
DEVINL v16h ldfrag(const _Float16* p) { return *reinterpret_cast<const v16h*>(p); }
DEVINL v8f wmma16(v16h a, v16h b, v8f c) {
  return __builtin_amdgcn_wmma_f32_16x16x32_f16(false, a, false, b, (short)0, c, false, false);
}
DEVINL float sigmoidf_(float x) { return 1.f / (1.f + __expf(-x)); }

// ---------------- weight packing into per-lane f16 fragments ----------------

// dilated-inception taps: 14 frags (p=0..6, mtile=0..1), branches k={2,3,6,7} end-aligned
__global__ void pack_tap_kernel(const float* w0, const float* w1, const float* w2,
                                const float* w3, _Float16* dst) {
  int idx = blockIdx.x * blockDim.x + threadIdx.x;
  if (idx >= 14 * 512) return;
  int f = idx >> 9, pos = idx & 511;
  int p = f >> 1, mt = f & 1;
  int lane = pos >> 4, h = pos & 15;
  int m = lane & 15, g = lane >> 4;
  int row = mt * 16 + m;
  int br = row >> 3, cl = row & 7;
  const int ks[4] = {2, 3, 6, 7};
  const float* w = (br == 0) ? w0 : (br == 1) ? w1 : (br == 2) ? w2 : w3;
  int k = ks[br];
  int j = p + k - 7;                         // tap p active iff p >= 7-k
  int cin = kmap(g, h);
  float v = (j >= 0) ? w[(cl * 32 + cin) * k + j] : 0.f;
  dst[idx] = (_Float16)v;
}

// time-sliced A frags for skip/skipE convs: src [M=64,32,T]; frag idx = t*4+mt
__global__ void pack_tslice_kernel(const float* w, _Float16* dst, int T, int Mtiles) {
  int idx = blockIdx.x * blockDim.x + threadIdx.x;
  if (idx >= T * Mtiles * 512) return;
  int f = idx >> 9, pos = idx & 511;
  int t = f / Mtiles, mt = f % Mtiles;
  int lane = pos >> 4, h = pos & 15;
  int m = lane & 15, g = lane >> 4;
  int row = mt * 16 + m;
  int cin = kmap(g, h);
  dst[idx] = (_Float16)w[(row * 32 + cin) * T + t];
}

// plain [rows,Ktot] matrices chunked K->32; frag idx = q*Mtiles+mt; zero row pad
__global__ void pack_mat_kernel(const float* w, _Float16* dst, int Ktot, int Mtiles,
                                int rows_valid) {
  int nfrags = (Ktot >> 5) * Mtiles;
  int idx = blockIdx.x * blockDim.x + threadIdx.x;
  if (idx >= nfrags * 512) return;
  int f = idx >> 9, pos = idx & 511;
  int q = f / Mtiles, mt = f % Mtiles;
  int lane = pos >> 4, h = pos & 15;
  int m = lane & 15, g = lane >> 4;
  int row = mt * 16 + m;
  int kg = q * 32 + kmap(g, h);
  float v = (row < rows_valid) ? w[row * Ktot + kg] : 0.f;
  dst[idx] = (_Float16)v;
}

// ---------------- start conv (1x1, K=2: VALU) ----------------
__global__ void start_kernel(const float* __restrict__ x, const float* __restrict__ w,
                             const float* __restrict__ bb, float* __restrict__ h) {
  size_t idx = (size_t)blockIdx.x * blockDim.x + threadIdx.x;
  if (idx >= (size_t)8 * 32 * 2000 * 64) return;
  int t = idx & 63;
  size_t r = idx >> 6;
  int n = r % 2000; r /= 2000;
  int c = r & 31;
  int b = (int)(r >> 5);
  float acc = bb[c] + w[c * 2 + 0] * x[(((size_t)b * 2 + 0) * 2000 + n) * 64 + t]
                    + w[c * 2 + 1] * x[(((size_t)b * 2 + 1) * 2000 + n) * 64 + t];
  h[idx] = acc;
}

// ---------------- skip0: [64 x (2*64)] GEMM per 16-node tile ----------------
__global__ void skip0_kernel(const float* __restrict__ x, const _Float16* __restrict__ fragA,
                             const float* __restrict__ sbias, float* __restrict__ skip) {
  int b = blockIdx.y, nt = blockIdx.x;
  int lane = threadIdx.x & 31, g = lane >> 4, nl = lane & 15;
  int node = nt * 16 + nl;
  v8f acc[4];
#pragma unroll
  for (int s = 0; s < 4; s++)
#pragma unroll
    for (int r = 0; r < 8; r++) acc[s][r] = 0.f;
#pragma unroll
  for (int q = 0; q < 4; q++) {
    const int cin = q >> 1;                       // K order: k = cin*64 + t
    const float* xb = &x[(((size_t)b * 2 + cin) * 2000 + node) * 64 + (q & 1) * 32];
    v16h bf;
#pragma unroll
    for (int h = 0; h < 16; h++) bf[h] = (_Float16)xb[kmap(g, h)];
#pragma unroll
    for (int s = 0; s < 4; s++)
      acc[s] = wmma16(ldfrag(fragA + ((q * 4 + s) * 512 + lane * 16)), bf, acc[s]);
  }
#pragma unroll
  for (int s = 0; s < 4; s++)
#pragma unroll
    for (int r = 0; r < 8; r++) {
      int sc = 16 * s + 8 * g + r;
      skip[((size_t)b * 64 + sc) * 2000 + node] = acc[s][r] + sbias[sc];
    }
}

// ---------------- main temporal-conv layer ----------------
__global__ void __launch_bounds__(128)
layer_kernel(const float* __restrict__ hin, float* __restrict__ hout,
             float* __restrict__ skip, float* __restrict__ stats,
             const _Float16* __restrict__ fF, const _Float16* __restrict__ fG,
             const _Float16* __restrict__ fS, const _Float16* __restrict__ fR,
             const float* __restrict__ fb, const float* __restrict__ gb,
             const float* __restrict__ sb, const float* __restrict__ rb,
             int Tin, int Tout, int dil) {
  __shared__ __align__(32) _Float16 hs[64 * 16 * 32];  // [t][node][slot] permuted, 64KB
  int b = blockIdx.y, nt = blockIdx.x;
  int tid = threadIdx.x, lane = tid & 31, wv = tid >> 5;
  int g = lane >> 4, nl = lane & 15;
  int node0 = nt * 16;

  __builtin_prefetch(&hin[(((size_t)b * 32) * 2000 + node0) * (size_t)Tin], 0, 1);
  // stage: each thread streams a contiguous time-row of one (channel,node) pair
  for (int base = tid; base < 512; base += 128) {
    int n = base >> 5, s = base & 31;
    int c = chperm(s);
    const float* src = &hin[(((size_t)b * 32 + c) * 2000 + node0 + n) * (size_t)Tin];
    for (int t = 0; t < Tin; t++)
      hs[(t * 16 + n) * 32 + s] = (_Float16)src[t];
  }
  __syncthreads();

  // loop-invariant bias fragments
  v8f bF0, bF1, bG0, bG1, bR0, bR1;
#pragma unroll
  for (int r = 0; r < 8; r++) {
    bF0[r] = fb[8 * g + r];       bF1[r] = fb[16 + 8 * g + r];
    bG0[r] = gb[8 * g + r];       bG1[r] = gb[16 + 8 * g + r];
    bR0[r] = rb[8 * g + r];       bR1[r] = rb[16 + 8 * g + r];
  }

  v8f sk[4];
#pragma unroll
  for (int s = 0; s < 4; s++)
#pragma unroll
    for (int r = 0; r < 8; r++) sk[s][r] = 0.f;
  float ls1 = 0.f, ls2 = 0.f;

  for (int t = wv; t < Tout; t += 4) {        // uniform per wave: EXEC all ones at WMMA
    v8f aF0 = bF0, aF1 = bF1, aG0 = bG0, aG1 = bG1;
#pragma unroll
    for (int p = 0; p < 7; p++) {
      const v16h bp = *reinterpret_cast<const v16h*>(
          &hs[(((t + p * dil) * 16 + nl) * 32) + 16 * g]);
      aF0 = wmma16(ldfrag(fF + ((p * 2 + 0) * 512 + lane * 16)), bp, aF0);
      aF1 = wmma16(ldfrag(fF + ((p * 2 + 1) * 512 + lane * 16)), bp, aF1);
      aG0 = wmma16(ldfrag(fG + ((p * 2 + 0) * 512 + lane * 16)), bp, aG0);
      aG1 = wmma16(ldfrag(fG + ((p * 2 + 1) * 512 + lane * 16)), bp, aG1);
    }
    // tanh * sigmoid -> register-chained B fragment for skip/res GEMMs
    v16h bh;
#pragma unroll
    for (int r = 0; r < 8; r++) {
      bh[r]     = (_Float16)(tanhf(aF0[r]) * sigmoidf_(aG0[r]));
      bh[8 + r] = (_Float16)(tanhf(aF1[r]) * sigmoidf_(aG1[r]));
    }
#pragma unroll
    for (int s = 0; s < 4; s++)
      sk[s] = wmma16(ldfrag(fS + ((t * 4 + s) * 512 + lane * 16)), bh, sk[s]);

    v8f r0 = bR0, r1 = bR1;
    r0 = wmma16(ldfrag(fR + (0 * 512 + lane * 16)), bh, r0);
    r1 = wmma16(ldfrag(fR + (1 * 512 + lane * 16)), bh, r1);

    // residual: the lane's 16 residual channels are exactly its B-fragment halves
    int toff = t + (Tin - Tout);               // residual end-aligned
    const v16h rfrag = *reinterpret_cast<const v16h*>(
        &hs[((toff * 16 + nl) * 32) + 16 * g]);
#pragma unroll
    for (int r = 0; r < 8; r++) {
      int c0 = 8 * g + r;
      float v0 = r0[r] + (float)rfrag[r];
      hout[(((size_t)b * 32 + c0) * 2000 + node0 + nl) * (size_t)Tout + t] = v0;
      ls1 += v0; ls2 += v0 * v0;
      int c1 = 16 + 8 * g + r;
      float v1 = r1[r] + (float)rfrag[8 + r];
      hout[(((size_t)b * 32 + c1) * 2000 + node0 + nl) * (size_t)Tout + t] = v1;
      ls1 += v1; ls2 += v1 * v1;
    }
  }
  // LayerNorm stats: wave reduce then 2 global atomics per wave
#pragma unroll
  for (int o = 16; o > 0; o >>= 1) {
    ls1 += __shfl_xor(ls1, o, 32);
    ls2 += __shfl_xor(ls2, o, 32);
  }
  if (lane == 0) { atomicAdd(&stats[b * 2 + 0], ls1); atomicAdd(&stats[b * 2 + 1], ls2); }
  // skip accumulation (bias added once by wave 0)
#pragma unroll
  for (int s = 0; s < 4; s++)
#pragma unroll
    for (int r = 0; r < 8; r++) {
      int sc = 16 * s + 8 * g + r;
      float v = sk[s][r] + ((wv == 0) ? sb[sc] : 0.f);
      atomicAdd(&skip[((size_t)b * 64 + sc) * 2000 + node0 + nl], v);
    }
}

// ---------------- LayerNorm apply ----------------
__global__ void ln_apply_kernel(float* __restrict__ h, const float* __restrict__ stats,
                                const float* __restrict__ nw, const float* __restrict__ nb,
                                int Tt) {
  size_t per = (size_t)32 * 2000 * Tt;
  size_t idx = (size_t)blockIdx.x * blockDim.x + threadIdx.x;
  if (idx >= (size_t)8 * per) return;
  int b = (int)(idx / per);
  size_t rem = idx - (size_t)b * per;
  float cnt = (float)per;
  float mu = stats[b * 2 + 0] / cnt;
  float var = stats[b * 2 + 1] / cnt - mu * mu;
  float rs = rsqrtf(var + 1e-5f);
  h[idx] = (h[idx] - mu) * rs * nw[rem] + nb[rem];
}

// ---------------- fused skipE + relu + end1 + relu + end2 ----------------
__global__ void final_kernel(const float* __restrict__ h, const float* __restrict__ skip,
                             const _Float16* __restrict__ fSE, const _Float16* __restrict__ fE1,
                             const _Float16* __restrict__ fE2, const float* __restrict__ seb,
                             const float* __restrict__ e1b, const float* __restrict__ e2b,
                             float* __restrict__ out) {
  __shared__ __align__(32) _Float16 hs[22 * 16 * 32];
  int b = blockIdx.y, nt = blockIdx.x;
  int lane = threadIdx.x & 31, g = lane >> 4, nl = lane & 15;
  int node0 = nt * 16;
  for (int base = lane; base < 512; base += 32) {
    int n = base >> 5, s = base & 31;
    int c = chperm(s);
    const float* src = &h[(((size_t)b * 32 + c) * 2000 + node0 + n) * 22];
#pragma unroll
    for (int t = 0; t < 22; t++)
      hs[(t * 16 + n) * 32 + s] = (_Float16)src[t];
  }
  __syncthreads();

  v8f sk[4];
#pragma unroll
  for (int s = 0; s < 4; s++)
#pragma unroll
    for (int r = 0; r < 8; r++)
      sk[s][r] = skip[((size_t)b * 64 + (16 * s + 8 * g + r)) * 2000 + node0 + nl];
#pragma unroll
  for (int t = 0; t < 22; t++) {
    const v16h bh = *reinterpret_cast<const v16h*>(&hs[((t * 16 + nl) * 32) + 16 * g]);
#pragma unroll
    for (int s = 0; s < 4; s++)
      sk[s] = wmma16(ldfrag(fSE + ((t * 4 + s) * 512 + lane * 16)), bh, sk[s]);
  }
#pragma unroll
  for (int s = 0; s < 4; s++)
#pragma unroll
    for (int r = 0; r < 8; r++)
      sk[s][r] = fmaxf(sk[s][r] + seb[16 * s + 8 * g + r], 0.f);

  v8f e1[8];
#pragma unroll
  for (int mt = 0; mt < 8; mt++)
#pragma unroll
    for (int r = 0; r < 8; r++) e1[mt][r] = e1b[16 * mt + 8 * g + r];
#pragma unroll
  for (int q = 0; q < 2; q++) {
    v16h bq;  // D->B register chaining from relu'd skip tiles (2q, 2q+1)
#pragma unroll
    for (int hh = 0; hh < 8; hh++) {
      bq[hh]     = (_Float16)sk[2 * q][hh];
      bq[8 + hh] = (_Float16)sk[2 * q + 1][hh];
    }
#pragma unroll
    for (int mt = 0; mt < 8; mt++)
      e1[mt] = wmma16(ldfrag(fE1 + ((q * 8 + mt) * 512 + lane * 16)), bq, e1[mt]);
  }
#pragma unroll
  for (int mt = 0; mt < 8; mt++)
#pragma unroll
    for (int r = 0; r < 8; r++) e1[mt][r] = fmaxf(e1[mt][r], 0.f);

  v8f e2;
#pragma unroll
  for (int r = 0; r < 8; r++) { int o = 8 * g + r; e2[r] = (o < 12) ? e2b[o] : 0.f; }
#pragma unroll
  for (int q = 0; q < 4; q++) {
    v16h bq;
#pragma unroll
    for (int hh = 0; hh < 8; hh++) {
      bq[hh]     = (_Float16)e1[2 * q][hh];
      bq[8 + hh] = (_Float16)e1[2 * q + 1][hh];
    }
    e2 = wmma16(ldfrag(fE2 + (q * 512 + lane * 16)), bq, e2);
  }
#pragma unroll
  for (int r = 0; r < 8; r++) {
    int o = 8 * g + r;
    if (o < 12) out[((size_t)b * 12 + o) * 2000 + node0 + nl] = e2[r];
  }
}

// ---------------- host orchestration ----------------
extern "C" void kernel_launch(void* const* d_in, const int* in_sizes, int n_in,
                              void* d_out, int out_size, void* d_ws, size_t ws_size,
                              hipStream_t stream) {
  (void)in_sizes; (void)n_in; (void)out_size; (void)ws_size;
  constexpr int B = 8, N = 2000;
  const int TOUT[3] = {58, 46, 22};
  const int TIN[3]  = {64, 58, 46};
  const int DIL[3]  = {1, 2, 4};

  char* ws = (char*)d_ws;
  size_t off = 0;
  auto take = [&](size_t bytes) { size_t o = off; off += (bytes + 255) & ~(size_t)255; return o; };
  const size_t oHA  = take((size_t)B * 32 * N * 64 * 4);
  const size_t oHB  = take((size_t)B * 32 * N * 64 * 4);
  const size_t oSK  = take((size_t)B * 64 * N * 4);
  const size_t oST  = take(B * 2 * 4);
  const size_t oFF  = take(3 * 14 * 512 * 2);
  const size_t oFG  = take(3 * 14 * 512 * 2);
  const size_t oFS  = take((58 + 46 + 22) * 4 * 512 * 2);
  const size_t oFR  = take(3 * 2 * 512 * 2);
  const size_t oFS0 = take(16 * 512 * 2);
  const size_t oFSE = take(22 * 4 * 512 * 2);
  const size_t oFE1 = take(16 * 512 * 2);
  const size_t oFE2 = take(4 * 512 * 2);
  const size_t oFB  = take(3 * 32 * 4);
  const size_t oGB  = take(3 * 32 * 4);

  _Float16* fF  = (_Float16*)(ws + oFF);
  _Float16* fG  = (_Float16*)(ws + oFG);
  _Float16* fS  = (_Float16*)(ws + oFS);
  _Float16* fR  = (_Float16*)(ws + oFR);
  _Float16* fS0 = (_Float16*)(ws + oFS0);
  _Float16* fSE = (_Float16*)(ws + oFSE);
  _Float16* fE1 = (_Float16*)(ws + oFE1);
  _Float16* fE2 = (_Float16*)(ws + oFE2);
  float* hA = (float*)(ws + oHA);
  float* hB = (float*)(ws + oHB);
  float* skip = (float*)(ws + oSK);
  float* stats = (float*)(ws + oST);
  float* fb = (float*)(ws + oFB);
  float* gb = (float*)(ws + oGB);
  const int soff[3] = {0, 58 * 4, (58 + 46) * 4};  // skip frag layer offsets

  // weight packing
  for (int i = 0; i < 3; i++) {
    pack_tap_kernel<<<28, 256, 0, stream>>>((const float*)d_in[7 + i * 4 + 0],
                                            (const float*)d_in[7 + i * 4 + 1],
                                            (const float*)d_in[7 + i * 4 + 2],
                                            (const float*)d_in[7 + i * 4 + 3],
                                            fF + (size_t)i * 14 * 512);
    pack_tap_kernel<<<28, 256, 0, stream>>>((const float*)d_in[31 + i * 4 + 0],
                                            (const float*)d_in[31 + i * 4 + 1],
                                            (const float*)d_in[31 + i * 4 + 2],
                                            (const float*)d_in[31 + i * 4 + 3],
                                            fG + (size_t)i * 14 * 512);
    pack_tslice_kernel<<<(TOUT[i] * 4 * 512 + 255) / 256, 256, 0, stream>>>(
        (const float*)d_in[61 + i], fS + (size_t)soff[i] * 512, TOUT[i], 4);
    pack_mat_kernel<<<4, 256, 0, stream>>>((const float*)d_in[55 + i],
                                           fR + (size_t)i * 2 * 512, 32, 2, 32);
  }
  pack_mat_kernel<<<32, 256, 0, stream>>>((const float*)d_in[5], fS0, 128, 4, 64);
  pack_tslice_kernel<<<(22 * 4 * 512 + 255) / 256, 256, 0, stream>>>(
      (const float*)d_in[73], fSE, 22, 4);
  pack_mat_kernel<<<32, 256, 0, stream>>>((const float*)d_in[75], fE1, 64, 8, 128);
  pack_mat_kernel<<<8, 256, 0, stream>>>((const float*)d_in[77], fE2, 128, 1, 12);

  // concat the 4 per-branch bias vectors into [32] per layer (filter, gate)
  for (int i = 0; i < 3; i++)
    for (int br = 0; br < 4; br++) {
      hipMemcpyAsync(fb + i * 32 + br * 8, d_in[19 + i * 4 + br], 8 * sizeof(float),
                     hipMemcpyDeviceToDevice, stream);
      hipMemcpyAsync(gb + i * 32 + br * 8, d_in[43 + i * 4 + br], 8 * sizeof(float),
                     hipMemcpyDeviceToDevice, stream);
    }

  // start 1x1 conv + skip0
  start_kernel<<<128000, 256, 0, stream>>>((const float*)d_in[0], (const float*)d_in[3],
                                           (const float*)d_in[4], hA);
  skip0_kernel<<<dim3(125, B), 32, 0, stream>>>((const float*)d_in[0], fS0,
                                                (const float*)d_in[6], skip);

  // three temporal-conv layers (ping-pong hA/hB)
  float* hin = hA; float* hout = hB;
  for (int i = 0; i < 3; i++) {
    hipMemsetAsync(stats, 0, B * 2 * sizeof(float), stream);
    layer_kernel<<<dim3(125, B), 128, 0, stream>>>(
        hin, hout, skip, stats, fF + (size_t)i * 14 * 512, fG + (size_t)i * 14 * 512,
        fS + (size_t)soff[i] * 512, fR + (size_t)i * 2 * 512, fb + i * 32, gb + i * 32,
        (const float*)d_in[64 + i], (const float*)d_in[58 + i], TIN[i], TOUT[i], DIL[i]);
    size_t tot = (size_t)B * 32 * N * TOUT[i];
    ln_apply_kernel<<<(unsigned)((tot + 255) / 256), 256, 0, stream>>>(
        hout, stats, (const float*)d_in[67 + i], (const float*)d_in[70 + i], TOUT[i]);
    float* tmp = hin; hin = hout; hout = tmp;
  }

  // hin now points at layer-2 output (hB), T=22
  final_kernel<<<dim3(125, B), 32, 0, stream>>>(
      hin, skip, fSE, fE1, fE2, (const float*)d_in[74], (const float*)d_in[76],
      (const float*)d_in[78], (float*)d_out);
}
